// ChannelModel_446676599302
// MI455X (gfx1250) — matrix operand order repaired and checked by reading, not compiled, not run on hardware
//
#include <hip/hip_runtime.h>
#include <hip/hip_bf16.h>
#include <math.h>

// ---------------------------------------------------------------------------
// Problem dims (fixed by the reference)
//   D: [16384, 2048] f32,  S: [1024, 2048] f32,  W: [2048, 2048] f32
// ---------------------------------------------------------------------------
#define NROW 16384
#define MROW 1024
#define DDIM 2048

typedef __attribute__((ext_vector_type(8)))  __bf16 bfx8;
typedef __attribute__((ext_vector_type(16))) __bf16 bfx16;
typedef __attribute__((ext_vector_type(8)))  float  fx8;
typedef int v4i __attribute__((vector_size(16)));   // exact pointee type of async builtin

// ---------------------------------------------------------------------------
// Async global->LDS staging (gfx1250 GLOBAL_LOAD_ASYNC_TO_LDS_B128, ASYNCcnt)
// with a synchronous fallback if the builtin is absent on this toolchain.
// Probe-confirmed signature: (AS1 v4i* src, AS3 v4i* dst, imm offset, imm cpol)
// ---------------------------------------------------------------------------
#if defined(__HIP_DEVICE_COMPILE__) && __has_builtin(__builtin_amdgcn_global_load_async_to_lds_b128)
#define ASYNC_LDS 1
#else
#define ASYNC_LDS 0
#endif

__device__ __forceinline__ void cp16(const __bf16* __restrict__ g, __bf16* __restrict__ l) {
#if ASYNC_LDS
    // generic LDS pointer low 32 bits == LDS offset (ISA 10.2 aperture rules)
    __builtin_amdgcn_global_load_async_to_lds_b128(
        (__attribute__((address_space(1))) v4i*)(uintptr_t)g,
        (__attribute__((address_space(3))) v4i*)(unsigned)(uintptr_t)l,
        0, 0);
#else
    *(bfx8*)l = *(const bfx8*)g;
#endif
}

__device__ __forceinline__ void wait_stage() {
#if ASYNC_LDS
#if __has_builtin(__builtin_amdgcn_s_wait_asynccnt)
    __builtin_amdgcn_s_wait_asynccnt(0);
#else
    asm volatile("s_wait_asynccnt 0" ::: "memory");
#endif
#endif
}

// ---------------------------------------------------------------------------
// Prep kernels
// ---------------------------------------------------------------------------
__global__ void cvt_f32_bf16(const float* __restrict__ in, __bf16* __restrict__ out, long n) {
    long i = (long)blockIdx.x * blockDim.x + threadIdx.x;
    long stride = (long)gridDim.x * blockDim.x;
    for (; i < n; i += stride) out[i] = (__bf16)in[i];
}

// S[rows][cols] (f32) -> ST[cols][rows] (bf16)
__global__ void transpose_f32_bf16(const float* __restrict__ S, __bf16* __restrict__ ST,
                                   int rows, int cols) {
    long n = (long)rows * cols;
    long i = (long)blockIdx.x * blockDim.x + threadIdx.x;
    long stride = (long)gridDim.x * blockDim.x;
    for (; i < n; i += stride) {
        int r = (int)(i / cols);
        int c = (int)(i % cols);
        ST[(size_t)c * rows + r] = (__bf16)S[i];
    }
}

__global__ void zero_f32(float* __restrict__ p, int n) {
    int i = blockIdx.x * blockDim.x + threadIdx.x;
    int stride = gridDim.x * blockDim.x;
    for (; i < n; i += stride) p[i] = 0.0f;
}

// ---------------------------------------------------------------------------
// WMMA GEMM:  C[M][N] = A[M][K] * B[N][K]^T   (A, B bf16 row-major, K-major)
// Block tile 128x256, BK=32, 256 threads = 8 waves in 2(M) x 4(N), wave tile
// 64x64 = 4x4 v_wmma_f32_16x16x32_bf16 tiles.  Double-buffered LDS panels,
// async staging, one barrier per K-step.
// MODE 0: store f32 to Cf        (scores)
// MODE 1: store bf16 to Cb       (att_S)
// MODE 2: logits[i] += sum_n C[i][n]*Dm[i][n]  (fused bilinear dot)
// ---------------------------------------------------------------------------
constexpr int BM = 128, BN = 256, BK = 32;
constexpr int LDP = 40;   // LDS row pitch in bf16 (+8 pad: 80B rows, 16B aligned)

__device__ __forceinline__ bfx16 ld_frag(const __bf16* __restrict__ p, int o0, int o1) {
    bfx8 lo = *(const bfx8*)(p + o0);
    bfx8 hi = *(const bfx8*)(p + o1);
    bfx16 r;
#pragma unroll
    for (int i = 0; i < 8; ++i) { r[i] = lo[i]; r[8 + i] = hi[i]; }
    return r;
}

template <int MODE>
__global__ __launch_bounds__(256)
void gemm_bt_kernel(const __bf16* __restrict__ A, const __bf16* __restrict__ B,
                    int M, int N, int K,
                    float* __restrict__ Cf, __bf16* __restrict__ Cb,
                    const float* __restrict__ Dm, float* __restrict__ logits) {
    __shared__ __bf16 Abuf[2][BM * LDP];   // 2*128*40*2 = 20480 B
    __shared__ __bf16 Bbuf[2][BN * LDP];   // 2*256*40*2 = 40960 B

    const int tid   = threadIdx.x;
    const int lane  = tid & 31;
    const int wave  = tid >> 5;
    const int waveM = wave & 1;   // 0..1  (M offset *64)
    const int waveN = wave >> 1;  // 0..3  (N offset *64)
    const int half  = lane >> 4;  // 0/1
    const int l15   = lane & 15;

    const int bm = blockIdx.y * BM;
    const int bn = blockIdx.x * BN;

    fx8 acc[4][4];
#pragma unroll
    for (int mt = 0; mt < 4; ++mt)
#pragma unroll
        for (int nt = 0; nt < 4; ++nt)
#pragma unroll
            for (int e = 0; e < 8; ++e) acc[mt][nt][e] = 0.0f;

    // Stage one 128x32 (A) + 256x32 (B) panel pair into LDS buffer `dst`.
    auto stage = [&](int dst, int k0) {
#pragma unroll
        for (int i = 0; i < 2; ++i) {            // A: 512 chunks / 256 thr
            int idx = tid + (i << 8);
            int row = idx >> 2;
            int cc  = (idx & 3) << 3;
            cp16(&A[(size_t)(bm + row) * K + k0 + cc], &Abuf[dst][row * LDP + cc]);
        }
#pragma unroll
        for (int i = 0; i < 4; ++i) {            // B: 1024 chunks / 256 thr
            int idx = tid + (i << 8);
            int row = idx >> 2;
            int cc  = (idx & 3) << 3;
            cp16(&B[(size_t)(bn + row) * K + k0 + cc], &Bbuf[dst][row * LDP + cc]);
        }
    };

    const int kTiles = K >> 5;
    stage(0, 0);
    wait_stage();
    __syncthreads();

    int buf = 0;
    for (int kt = 0; kt < kTiles; ++kt) {
        if (kt + 1 < kTiles) stage(buf ^ 1, (kt + 1) << 5);

        // Fragment gathers per ISA 7.12.2 (16-bit A 16x32, B 32x16 layouts)
        bfx16 afrag[4];
#pragma unroll
        for (int mt = 0; mt < 4; ++mt)
            afrag[mt] = ld_frag(&Abuf[buf][(waveM * 64 + mt * 16 + l15) * LDP],
                                half * 8, 16 + half * 8);
#pragma unroll
        for (int nt = 0; nt < 4; ++nt) {
            bfx16 bfrag = ld_frag(&Bbuf[buf][(waveN * 64 + nt * 16 + l15) * LDP],
                                  half * 8, 16 + half * 8);
#pragma unroll
            for (int mt = 0; mt < 4; ++mt)
                acc[mt][nt] = __builtin_amdgcn_wmma_f32_16x16x32_bf16(
                    false, afrag[mt], false, bfrag,
                    (short)0, acc[mt][nt], false, false);
        }

        wait_stage();       // next-buffer asyncs complete (this wave)
        __syncthreads();    // all waves: staging visible, reads of `buf` done
        buf ^= 1;
    }

    // Epilogue. C/D layout: VGPR v, lane L -> row = v + 8*(L>>4), col = L&15.
#pragma unroll
    for (int mt = 0; mt < 4; ++mt) {
#pragma unroll
        for (int nt = 0; nt < 4; ++nt) {
            const int gmBase = bm + waveM * 64 + mt * 16 + half * 8;
            const int gn     = bn + waveN * 64 + nt * 16 + l15;
#pragma unroll
            for (int v = 0; v < 8; ++v) {
                const int gm = gmBase + v;
                float val = acc[mt][nt][v];
                if constexpr (MODE == 0) {
                    Cf[(size_t)gm * N + gn] = val;
                } else if constexpr (MODE == 1) {
                    Cb[(size_t)gm * N + gn] = (__bf16)val;
                } else {
                    float p = val * Dm[(size_t)gm * N + gn];
                    p += __shfl_xor(p, 1, 32);
                    p += __shfl_xor(p, 2, 32);
                    p += __shfl_xor(p, 4, 32);
                    p += __shfl_xor(p, 8, 32);
                    if (l15 == 0) atomicAdd(&logits[gm], p);
                }
            }
        }
    }
}

// ---------------------------------------------------------------------------
// Row softmax: att[r][c] = softmax(scores[r][:] * scale), output bf16
// ---------------------------------------------------------------------------
__global__ __launch_bounds__(256)
void softmax_rows(const float* __restrict__ scores, __bf16* __restrict__ att,
                  int cols, float scale) {
    __shared__ float red[256];
    const int row = blockIdx.x;
    const float* s = scores + (size_t)row * cols;

    float lmax = -INFINITY;
    for (int c = threadIdx.x; c < cols; c += 256) lmax = fmaxf(lmax, s[c] * scale);
    red[threadIdx.x] = lmax;
    __syncthreads();
    for (int off = 128; off; off >>= 1) {
        if (threadIdx.x < off)
            red[threadIdx.x] = fmaxf(red[threadIdx.x], red[threadIdx.x + off]);
        __syncthreads();
    }
    const float m = red[0];
    __syncthreads();

    float lsum = 0.0f;
    for (int c = threadIdx.x; c < cols; c += 256) lsum += __expf(s[c] * scale - m);
    red[threadIdx.x] = lsum;
    __syncthreads();
    for (int off = 128; off; off >>= 1) {
        if (threadIdx.x < off) red[threadIdx.x] += red[threadIdx.x + off];
        __syncthreads();
    }
    const float inv = 1.0f / red[0];

    for (int c = threadIdx.x; c < cols; c += 256)
        att[(size_t)row * cols + c] = (__bf16)(__expf(s[c] * scale - m) * inv);
}

// ---------------------------------------------------------------------------
// out[0] = sum_i log_sigmoid(logits[i]), numerically stable
// ---------------------------------------------------------------------------
__global__ __launch_bounds__(256)
void logsig_sum(const float* __restrict__ logits, int n, float* __restrict__ out) {
    __shared__ float red[256];
    float acc = 0.0f;
    for (int i = threadIdx.x; i < n; i += 256) {
        float x = logits[i];
        acc += fminf(x, 0.0f) - log1pf(__expf(-fabsf(x)));
    }
    red[threadIdx.x] = acc;
    __syncthreads();
    for (int off = 128; off; off >>= 1) {
        if (threadIdx.x < off) red[threadIdx.x] += red[threadIdx.x + off];
        __syncthreads();
    }
    if (threadIdx.x == 0) out[0] = red[0];
}

// ---------------------------------------------------------------------------
// Launch: prep -> GEMM1(scores) -> softmax -> GEMM2(att_S) -> GEMM3+dot -> sum
// ---------------------------------------------------------------------------
extern "C" void kernel_launch(void* const* d_in, const int* in_sizes, int n_in,
                              void* d_out, int out_size, void* d_ws, size_t ws_size,
                              hipStream_t stream) {
    (void)in_sizes; (void)n_in; (void)out_size; (void)ws_size;
    const float* D = (const float*)d_in[0];
    const float* S = (const float*)d_in[1];
    const float* W = (const float*)d_in[2];
    float* out = (float*)d_out;

    // Workspace layout (bytes). attS_bf overlays scores (scores dead after softmax).
    char* ws = (char*)d_ws;
    __bf16* Dbf    = (__bf16*)(ws);                           // 16384*2048*2 = 67108864
    __bf16* Sbf    = (__bf16*)(ws + 67108864UL);              //  1024*2048*2 =  4194304
    __bf16* Wbf    = (__bf16*)(ws + 71303168UL);              //  2048*2048*2 =  8388608
    __bf16* STbf   = (__bf16*)(ws + 79691776UL);              //  2048*1024*2 =  4194304
    float*  scores = (float*) (ws + 83886080UL);              // 16384*1024*4 = 67108864
    __bf16* attSbf = (__bf16*)(ws + 83886080UL);              // overlays scores (same size)
    __bf16* attbf  = (__bf16*)(ws + 150994944UL);             // 16384*1024*2 = 33554432
    float*  logits = (float*) (ws + 184549376UL);             // 16384*4
    // total: ~184.6 MB

    // Prep: bf16 conversions, S transpose, zero logits
    cvt_f32_bf16<<<4096, 256, 0, stream>>>(D, Dbf, (long)NROW * DDIM);
    cvt_f32_bf16<<<1024, 256, 0, stream>>>(S, Sbf, (long)MROW * DDIM);
    cvt_f32_bf16<<<2048, 256, 0, stream>>>(W, Wbf, (long)DDIM * DDIM);
    transpose_f32_bf16<<<1024, 256, 0, stream>>>(S, STbf, MROW, DDIM);
    zero_f32<<<64, 256, 0, stream>>>(logits, NROW);

    // GEMM1: scores[n,m] = Dbf @ Sbf^T   (scale applied inside softmax)
    gemm_bt_kernel<0><<<dim3(MROW / BN, NROW / BM), 256, 0, stream>>>(
        Dbf, Sbf, NROW, MROW, DDIM, scores, nullptr, nullptr, nullptr);

    // Softmax rows -> bf16 attention weights
    const float scale = 1.0f / sqrtf((float)DDIM);
    softmax_rows<<<NROW, 256, 0, stream>>>(scores, attbf, MROW, scale);

    // GEMM2: att_S[n,e] = attbf @ STbf^T   (K = 1024)
    gemm_bt_kernel<1><<<dim3(DDIM / BN, NROW / BM), 256, 0, stream>>>(
        attbf, STbf, NROW, DDIM, MROW, nullptr, attSbf, nullptr, nullptr);

    // GEMM3 + fused dot: logits[i] += sum_d (attS @ W^T)[i,d] * D[i,d]
    gemm_bt_kernel<2><<<dim3(DDIM / BN, NROW / BM), 256, 0, stream>>>(
        attSbf, Wbf, NROW, DDIM, DDIM, nullptr, nullptr, D, logits);

    // Final: sum log_sigmoid(logits)
    logsig_sum<<<1, 256, 0, stream>>>(logits, NROW, out);
}
